// TemporalDebertaAggregatorV2_20237885898763
// MI455X (gfx1250) — matrix operand order repaired and checked by reading, not compiled, not run on hardware
//
#include <hip/hip_runtime.h>
#include <hip/hip_bf16.h>

typedef unsigned short u16;
typedef __attribute__((ext_vector_type(16))) __bf16 v16bf;
typedef __attribute__((ext_vector_type(8)))  float  v8f;
typedef unsigned int uint32x4_t __attribute__((ext_vector_type(4)));
typedef int          int32x8_t  __attribute__((ext_vector_type(8)));
typedef int          int32x4_t  __attribute__((ext_vector_type(4)));

#define BB   32
#define SS   512
#define HD   128
#define NL   2
#define NH   4
#define DH   32
#define DFF  512
#define RR   1024
#define SPANC 512
#define KC   128   // K-chunk staged in LDS per GEMM step

#if defined(__has_builtin)
#if __has_builtin(__builtin_amdgcn_tensor_load_to_lds)
#define HAVE_TDM 1
#endif
#endif

__device__ __forceinline__ u16 f2bf(float f) {
    unsigned u = __float_as_uint(f);
    unsigned r = u + 0x7FFFu + ((u >> 16) & 1u);   // round-to-nearest-even
    return (u16)(r >> 16);
}

union BfVec { v16bf v; uint4 q[2]; u16 u[16]; };

// A-operand per-lane load: 16 contiguous bf16 of the 32-long K chunk
__device__ __forceinline__ v16bf load_a_tile(const u16* p) {
    BfVec r;
    r.q[0] = *(const uint4*)(p);
    r.q[1] = *(const uint4*)(p + 8);
    return r.v;
}
// B-operand per-lane load: 16 contiguous bf16
__device__ __forceinline__ v16bf load_b_tile(const u16* p) {
    BfVec r;
    r.q[0] = *(const uint4*)(p);
    r.q[1] = *(const uint4*)(p + 8);
    return r.v;
}

#define WMMA_BF16(a, b, c) \
    __builtin_amdgcn_wmma_f32_16x16x32_bf16(false, (a), false, (b), (short)0, (c), false, false)

// ---------------------------------------------------------------------------
// Prep: insert CLS at first_pad, build updated mask, convert hs to bf16
// ---------------------------------------------------------------------------
__global__ void prep_kernel(const float* __restrict__ x, const int* __restrict__ mask,
                            const int* __restrict__ last_attn, const float* __restrict__ cls,
                            float* __restrict__ hsF, u16* __restrict__ hsB,
                            float* __restrict__ mF) {
    int bs = blockIdx.x;
    int c  = threadIdx.x;
    int b  = bs >> 9;
    int s  = bs & (SS - 1);
    int fp = last_attn[b] + 1; if (fp > SS - 1) fp = SS - 1;
    float val = (s == fp) ? cls[c] : x[(size_t)bs * HD + c];
    hsF[(size_t)bs * HD + c] = val;
    hsB[(size_t)bs * HD + c] = f2bf(val);
    if (c == 0) mF[bs] = (mask[bs] != 0 || s == fp) ? 1.f : 0.f;
}

__global__ void conv_bf_kernel(const float* __restrict__ in, u16* __restrict__ out, int n) {
    int i = blockIdx.x * 256 + threadIdx.x;
    if (i < n) out[i] = f2bf(in[i]);
}

// out[n*K + k] = bf16(in[k*N + n])   (W row-major K x N  ->  W^T row-major N x K)
__global__ void transpose_bf_kernel(const float* __restrict__ in, u16* __restrict__ out,
                                    int K, int N) {
    int idx = blockIdx.x * 256 + threadIdx.x;
    if (idx >= K * N) return;
    int n = idx / K, k = idx % K;
    out[idx] = f2bf(in[(size_t)k * N + n]);
}

__global__ void gelu_bf_kernel(const float* __restrict__ in, u16* __restrict__ out, int n) {
    int i = blockIdx.x * 256 + threadIdx.x;
    if (i >= n) return;
    float v = in[i];
    float g = 0.5f * v * (1.f + erff(v * 0.70710678118f));
    out[i] = f2bf(g);
}

// ---------------------------------------------------------------------------
// bf16 GEMM, 64x64 block tile, 4 waves x (16 x 64).  Weight panel (64 rows of
// W^T x KC cols) staged into LDS with the Tensor Data Mover, consumed by all
// waves; A streamed from global.  vt_mode=1 stores bf16 output as (b,h,d,S).
// ---------------------------------------------------------------------------
__global__ __launch_bounds__(128)
void gemm_bf16_kernel(const u16* __restrict__ A, const u16* __restrict__ Wt,
                      const float* __restrict__ bias,
                      float* __restrict__ outF, u16* __restrict__ outB,
                      int M, int N, int K, int vt_mode) {
    __shared__ u16 bpanel[64 * KC];   // 16 KB staged weight panel

    int lane = threadIdx.x & 31;
    int warp = threadIdx.x >> 5;
    int nb = N >> 6;
    int bm = blockIdx.x / nb, bn = blockIdx.x % nb;
    int m0 = (bm << 6) + (warp << 4);
    int n0 = bn << 6;
    int hi = lane >> 4, l16 = lane & 15;

    v8f acc[4] = {{}, {}, {}, {}};
    const u16* arow = A + (size_t)(m0 + l16) * K + (hi ? 16 : 0);

    for (int k0 = 0; k0 < K; k0 += KC) {
#ifdef HAVE_TDM
        if (warp == 0) {
            // Tensor DMA descriptor: 2-D tile, rows [n0,n0+64) x cols [k0,k0+KC)
            unsigned long long ga = (unsigned long long)(uintptr_t)(Wt + (size_t)n0 * K + k0);
            unsigned lds_off = (unsigned)(uintptr_t)(&bpanel[0]);
            uint32x4_t g0;
            g0[0] = 1u;                                   // count=1 valid descriptor
            g0[1] = lds_off;                              // LDS destination
            g0[2] = (unsigned)(ga & 0xFFFFFFFFu);         // global addr lo
            g0[3] = (unsigned)((ga >> 32) & 0x1FFFFFFu) | (2u << 30);  // hi | type=2
            int32x8_t g1;
            g1[0] = 1 << 16;                              // data_size = 2 bytes
            g1[1] = (K & 0xFFFF) << 16;                   // tensor_dim0[15:0]
            g1[2] = (K >> 16) | ((N & 0xFFFF) << 16);     // tensor_dim0[31:16] | tensor_dim1[15:0]
            g1[3] = (N >> 16) | (KC << 16);               // tensor_dim1[31:16] | tile_dim0
            g1[4] = 64;                                   // tile_dim1 = 64 rows
            g1[5] = K;                                    // tensor_dim0_stride
            g1[6] = 0; g1[7] = 0;
            int32x4_t z4 = {0, 0, 0, 0};
#if __clang_major__ >= 23
            int32x8_t z8 = {0, 0, 0, 0, 0, 0, 0, 0};
            __builtin_amdgcn_tensor_load_to_lds(g0, g1, z4, z4, z8, 0);
#else
            __builtin_amdgcn_tensor_load_to_lds(g0, g1, z4, z4, 0);
#endif
            __builtin_amdgcn_s_wait_tensorcnt(0);
        }
        __syncthreads();
#else
        {
            const u16* src = Wt + (size_t)n0 * K + k0;
            for (int v = threadIdx.x; v < 64 * (KC / 8); v += 128) {
                int r = v / (KC / 8), c = v % (KC / 8);
                ((uint4*)bpanel)[r * (KC / 8) + c] = *(const uint4*)(src + (size_t)r * K + c * 8);
            }
            __syncthreads();
        }
#endif
        for (int kk = 0; kk < KC; kk += 32) {
            v16bf a = load_a_tile(arow + k0 + kk);
            #pragma unroll
            for (int t = 0; t < 4; ++t) {
                v16bf bt = load_b_tile(&bpanel[((t << 4) + l16) * KC + kk + (hi ? 16 : 0)]);
                acc[t] = WMMA_BF16(a, bt, acc[t]);
            }
        }
        __syncthreads();
    }

    for (int t = 0; t < 4; ++t) {
        for (int jj = 0; jj < 8; ++jj) {
            int m = m0 + jj + (hi << 3);
            int n = n0 + (t << 4) + l16;
            float va = acc[t][jj] + (bias ? bias[n] : 0.f);
            if (outF) outF[(size_t)m * N + n] = va;
            if (outB) {
                if (!vt_mode) {
                    outB[(size_t)m * N + n] = f2bf(va);
                } else {  // V stored transposed: (b, head, d, S)
                    int b = m / SS, s = m % SS;
                    int h1 = n / DH, d1 = n % DH;
                    outB[(((size_t)b * NH + h1) * DH + d1) * SS + s] = f2bf(va);
                }
            }
        }
    }
}

// ---------------------------------------------------------------------------
// Disentangled attention. One wave per (b, head, 16-row q tile).
// scores[q,k] = (q·k + q·pos_k[q-k+S] + k·pos_q[q-k+S]) / sqrt(3d)
// ---------------------------------------------------------------------------
__global__ __launch_bounds__(32)
void attention_kernel(const u16* __restrict__ qB, const u16* __restrict__ kB,
                      const u16* __restrict__ vT, const u16* __restrict__ poskB,
                      const u16* __restrict__ posqB, const float* __restrict__ mF,
                      u16* __restrict__ ctxB) {
    __shared__ float sc[16][SS];     // 32 KB score row block
    __shared__ u16   pb[16][SS];     // 16 KB bf16 probabilities
    __shared__ float gb[16][32];     // c2p band tile
    __shared__ float hb[16][32];     // p2c band tile

    int idb = blockIdx.x;
    int qt = idb & 31, hh = (idb >> 5) & 3, b = idb >> 7;
    int q0 = qt << 4;
    int lane = threadIdx.x & 31, hi = lane >> 4, l16 = lane & 15;
    const float rscale = 0.1020620726f;  // 1/sqrt(3*32)
    const float* mrow = mF + (size_t)b * SS;

    // q tile (A layout), valid per q row
    v16bf qa = load_a_tile(qB + ((size_t)(b * SS + q0 + l16)) * HD + hh * DH + (hi ? 16 : 0));
    float mq[8];
    for (int jj = 0; jj < 8; ++jj) mq[jj] = mrow[q0 + jj + (hi << 3)];

    for (int kt = 0; kt < 32; ++kt) {
        int k0 = kt << 4;
        if (kt < 31)  // prefetch next key tile (global_prefetch_b8)
            __builtin_prefetch(kB + ((size_t)(b * SS + k0 + 16 + l16)) * HD + hh * DH, 0, 0);

        // keys: B layout (columns = key positions) and A layout (rows = key positions)
        v16bf kb = load_b_tile(kB + ((size_t)(b * SS + k0 + l16)) * HD + hh * DH + (hi ? 16 : 0));
        v16bf ka = load_a_tile(kB + ((size_t)(b * SS + k0 + l16)) * HD + hh * DH + (hi ? 16 : 0));

        int rbase = q0 - k0 + SPANC - 15;
        int r0 = rbase + l16;      r0 = r0 < 0 ? 0 : (r0 > RR - 1 ? RR - 1 : r0);
        int r1 = rbase + 16 + l16; r1 = r1 < 0 ? 0 : (r1 > RR - 1 ? RR - 1 : r1);
        v16bf pk0 = load_b_tile(poskB + (size_t)r0 * HD + hh * DH + (hi ? 16 : 0));
        v16bf pk1 = load_b_tile(poskB + (size_t)r1 * HD + hh * DH + (hi ? 16 : 0));
        v16bf pq0 = load_b_tile(posqB + (size_t)r0 * HD + hh * DH + (hi ? 16 : 0));
        v16bf pq1 = load_b_tile(posqB + (size_t)r1 * HD + hh * DH + (hi ? 16 : 0));

        v8f c0 = {}, g0 = {}, g1 = {}, h0 = {}, h1 = {};
        c0 = WMMA_BF16(qa, kb,  c0);   // content-content
        g0 = WMMA_BF16(qa, pk0, g0);   // c2p band, r = rbase..rbase+15
        g1 = WMMA_BF16(qa, pk1, g1);   // c2p band, r = rbase+16..
        h0 = WMMA_BF16(ka, pq0, h0);   // p2c band
        h1 = WMMA_BF16(ka, pq1, h1);

        for (int jj = 0; jj < 8; ++jj) {
            int rsel = jj + (hi << 3);
            gb[rsel][l16] = g0[jj]; gb[rsel][16 + l16] = g1[jj];
            hb[rsel][l16] = h0[jj]; hb[rsel][16 + l16] = h1[jj];
        }
        __syncthreads();
        float kv = mrow[k0 + l16];
        for (int jj = 0; jj < 8; ++jj) {
            int i = jj + (hi << 3), j = l16;
            int idx = i - j + 15;                         // band offset, 0..30
            float s = (c0[jj] + gb[i][idx] + hb[j][idx]) * rscale;
            bool valid = (mq[jj] > 0.f) && (kv > 0.f);
            sc[i][k0 + j] = valid ? s : -1e30f;
        }
        __syncthreads();
    }

    // softmax: lane -> row l16, half hi (cols 0..255 / 256..511)
    {
        int row = l16, basec = hi ? 256 : 0;
        float mx = -3.4e38f;
        for (int c = 0; c < 256; ++c) mx = fmaxf(mx, sc[row][basec + c]);
        mx = fmaxf(mx, __shfl_xor(mx, 16, 32));
        float sum = 0.f;
        for (int c = 0; c < 256; ++c) {
            float v = sc[row][basec + c];
            float p = (v <= -1e29f) ? 0.f : __expf(v - mx);
            sc[row][basec + c] = p;
            sum += p;
        }
        sum += __shfl_xor(sum, 16, 32);
        float rs = (sum > 0.f) ? 1.f / sum : 0.f;
        for (int c = 0; c < 256; ++c) pb[row][basec + c] = f2bf(sc[row][basec + c] * rs);
    }
    __syncthreads();

    // ctx = probs(16x512) @ V(512x32), V read from transposed layout
    v8f a0 = {}, a1 = {};
    const u16* vrow0 = vT + (((size_t)(b * NH + hh)) * DH + l16) * SS + (hi ? 16 : 0);
    const u16* vrow1 = vT + (((size_t)(b * NH + hh)) * DH + 16 + l16) * SS + (hi ? 16 : 0);
    for (int k0 = 0; k0 < SS; k0 += 32) {
        v16bf pa  = load_a_tile(&pb[l16][k0 + (hi ? 16 : 0)]);
        v16bf vb0 = load_b_tile(vrow0 + k0);
        v16bf vb1 = load_b_tile(vrow1 + k0);
        a0 = WMMA_BF16(pa, vb0, a0);
        a1 = WMMA_BF16(pa, vb1, a1);
    }
    for (int jj = 0; jj < 8; ++jj) {
        int i = jj + (hi << 3);
        size_t o = ((size_t)(b * SS + q0 + i)) * HD + hh * DH;
        ctxB[o + l16]      = f2bf(a0[jj]);
        ctxB[o + 16 + l16] = f2bf(a1[jj]);
    }
}

// ---------------------------------------------------------------------------
// LayerNorm(z + res) * g + b  ->  hsF (f32) + hsB (bf16). One wave per row.
// ---------------------------------------------------------------------------
__global__ void ln_kernel(const float* zin, const float* res,
                          const float* __restrict__ g, const float* __restrict__ bb,
                          float* hsF, u16* __restrict__ hsB) {
    int lane = threadIdx.x & 31;
    int row  = blockIdx.x * (blockDim.x >> 5) + (threadIdx.x >> 5);
    const float* zi = zin + (size_t)row * HD;
    const float* rr = res + (size_t)row * HD;
    float v[4];
    float sum = 0.f;
    for (int j = 0; j < 4; ++j) { v[j] = zi[lane + 32 * j] + rr[lane + 32 * j]; sum += v[j]; }
    for (int o = 16; o >= 1; o >>= 1) sum += __shfl_xor(sum, o, 32);
    float mu = sum * (1.f / HD);
    float var = 0.f;
    for (int j = 0; j < 4; ++j) { float d = v[j] - mu; var += d * d; }
    for (int o = 16; o >= 1; o >>= 1) var += __shfl_xor(var, o, 32);
    float inv = rsqrtf(var * (1.f / HD) + 1e-7f);
    for (int j = 0; j < 4; ++j) {
        int c = lane + 32 * j;
        float o = (v[j] - mu) * inv * g[c] + bb[c];
        hsF[(size_t)row * HD + c] = o;
        hsB[(size_t)row * HD + c] = f2bf(o);
    }
}

__global__ void gather_kernel(const float* __restrict__ hsF, const int* __restrict__ last_attn,
                              float* __restrict__ out) {
    int i = blockIdx.x * 128 + threadIdx.x;
    if (i >= BB * HD) return;
    int b = i >> 7, c = i & 127;
    int fp = last_attn[b] + 1; if (fp > SS - 1) fp = SS - 1;
    out[i] = hsF[((size_t)b * SS + fp) * HD + c];
}

// ---------------------------------------------------------------------------
extern "C" void kernel_launch(void* const* d_in, const int* in_sizes, int n_in,
                              void* d_out, int out_size, void* d_ws, size_t ws_size,
                              hipStream_t stream) {
    (void)in_sizes; (void)n_in; (void)out_size; (void)ws_size;

    const float* x    = (const float*)d_in[0];
    const int*   mask = (const int*)d_in[1];
    const int*   last = (const int*)d_in[2];
    const float* cls  = (const float*)d_in[3];
    const float* rel  = (const float*)d_in[4];
    const float* Wq  = (const float*)d_in[5];  const float* bq  = (const float*)d_in[6];
    const float* Wk  = (const float*)d_in[7];  const float* bk  = (const float*)d_in[8];
    const float* Wv  = (const float*)d_in[9];  const float* bv  = (const float*)d_in[10];
    const float* Wpk = (const float*)d_in[11]; const float* bpk = (const float*)d_in[12];
    const float* Wpq = (const float*)d_in[13]; const float* bpq = (const float*)d_in[14];
    const float* Wao = (const float*)d_in[15]; const float* bao = (const float*)d_in[16];
    const float* g1  = (const float*)d_in[17]; const float* b1  = (const float*)d_in[18];
    const float* Wi  = (const float*)d_in[19]; const float* bi  = (const float*)d_in[20];
    const float* Wo  = (const float*)d_in[21]; const float* bo  = (const float*)d_in[22];
    const float* g2  = (const float*)d_in[23]; const float* b2  = (const float*)d_in[24];

    char* base = (char*)d_ws; size_t off = 0;
    auto alloc = [&](size_t bytes) -> void* {
        off = (off + 255) & ~(size_t)255;
        void* p = base + off; off += bytes; return p;
    };

    const int MR = BB * SS;  // 16384 rows
    float* hsF   = (float*)alloc((size_t)MR * HD * 4);
    u16*   hsB   = (u16*)  alloc((size_t)MR * HD * 2);
    float* mF    = (float*)alloc((size_t)MR * 4);
    u16*   relB  = (u16*)  alloc((size_t)RR * HD * 2);
    u16*   qBuf  = (u16*)  alloc((size_t)MR * HD * 2);
    u16*   kBuf  = (u16*)  alloc((size_t)MR * HD * 2);
    u16*   vTb   = (u16*)  alloc((size_t)MR * HD * 2);
    u16*   ctxB  = (u16*)  alloc((size_t)MR * HD * 2);
    u16*   poskB = (u16*)  alloc((size_t)RR * HD * 2);
    u16*   posqB = (u16*)  alloc((size_t)RR * HD * 2);
    float* tmpF  = (float*)alloc((size_t)MR * HD * 4);
    float* ff1F  = (float*)alloc((size_t)MR * DFF * 4);
    u16*   ffB   = (u16*)  alloc((size_t)MR * DFF * 2);

    u16 *WqT[NL], *WkT[NL], *WvT[NL], *WpkT[NL], *WpqT[NL], *WaoT[NL], *WiT[NL], *WoT[NL];
    for (int l = 0; l < NL; ++l) {
        WqT[l]  = (u16*)alloc(HD * HD * 2);
        WkT[l]  = (u16*)alloc(HD * HD * 2);
        WvT[l]  = (u16*)alloc(HD * HD * 2);
        WpkT[l] = (u16*)alloc(HD * HD * 2);
        WpqT[l] = (u16*)alloc(HD * HD * 2);
        WaoT[l] = (u16*)alloc(HD * HD * 2);
        WiT[l]  = (u16*)alloc(HD * DFF * 2);
        WoT[l]  = (u16*)alloc(DFF * HD * 2);
    }

    prep_kernel<<<MR, HD, 0, stream>>>(x, mask, last, cls, hsF, hsB, mF);
    conv_bf_kernel<<<(RR * HD + 255) / 256, 256, 0, stream>>>(rel, relB, RR * HD);

    const int hh2 = HD * HD;
    for (int l = 0; l < NL; ++l) {
        transpose_bf_kernel<<<(hh2 + 255) / 256, 256, 0, stream>>>(Wq  + l * hh2, WqT[l],  HD, HD);
        transpose_bf_kernel<<<(hh2 + 255) / 256, 256, 0, stream>>>(Wk  + l * hh2, WkT[l],  HD, HD);
        transpose_bf_kernel<<<(hh2 + 255) / 256, 256, 0, stream>>>(Wv  + l * hh2, WvT[l],  HD, HD);
        transpose_bf_kernel<<<(hh2 + 255) / 256, 256, 0, stream>>>(Wpk + l * hh2, WpkT[l], HD, HD);
        transpose_bf_kernel<<<(hh2 + 255) / 256, 256, 0, stream>>>(Wpq + l * hh2, WpqT[l], HD, HD);
        transpose_bf_kernel<<<(hh2 + 255) / 256, 256, 0, stream>>>(Wao + l * hh2, WaoT[l], HD, HD);
        transpose_bf_kernel<<<(HD * DFF + 255) / 256, 256, 0, stream>>>(Wi + (size_t)l * HD * DFF, WiT[l], HD, DFF);
        transpose_bf_kernel<<<(DFF * HD + 255) / 256, 256, 0, stream>>>(Wo + (size_t)l * DFF * HD, WoT[l], DFF, HD);
    }

    auto gemm = [&](const u16* A, const u16* Wt, const float* bias, float* oF, u16* oB,
                    int M, int N, int K, int vt) {
        int blocks = (M / 64) * (N / 64);
        gemm_bf16_kernel<<<blocks, 128, 0, stream>>>(A, Wt, bias, oF, oB, M, N, K, vt);
    };

    for (int l = 0; l < NL; ++l) {
        gemm(hsB,  WqT[l],  bq  + l * HD, nullptr, qBuf,  MR, HD, HD, 0);
        gemm(hsB,  WkT[l],  bk  + l * HD, nullptr, kBuf,  MR, HD, HD, 0);
        gemm(hsB,  WvT[l],  bv  + l * HD, nullptr, vTb,   MR, HD, HD, 1);
        gemm(relB, WpkT[l], bpk + l * HD, nullptr, poskB, RR, HD, HD, 0);
        gemm(relB, WpqT[l], bpq + l * HD, nullptr, posqB, RR, HD, HD, 0);

        attention_kernel<<<BB * NH * (SS / 16), 32, 0, stream>>>(qBuf, kBuf, vTb, poskB, posqB, mF, ctxB);

        gemm(ctxB, WaoT[l], bao + l * HD, tmpF, nullptr, MR, HD, HD, 0);
        ln_kernel<<<MR / 8, 256, 0, stream>>>(tmpF, hsF, g1 + l * HD, b1 + l * HD, hsF, hsB);

        gemm(hsB, WiT[l], bi + l * DFF, ff1F, nullptr, MR, DFF, HD, 0);
        gelu_bf_kernel<<<(MR * DFF + 255) / 256, 256, 0, stream>>>(ff1F, ffB, MR * DFF);
        gemm(ffB, WoT[l], bo + l * HD, tmpF, nullptr, MR, HD, DFF, 0);
        ln_kernel<<<MR / 8, 256, 0, stream>>>(tmpF, hsF, g2 + l * HD, b2 + l * HD, hsF, hsB);
    }

    gather_kernel<<<(BB * HD + 127) / 128, 128, 0, stream>>>(hsF, last, (float*)d_out);
}
